// DiTAttention_39651138076999
// MI455X (gfx1250) — compile-verified, matrix operand order checked by reading, not run on hardware
//
#include <hip/hip_runtime.h>

typedef __attribute__((ext_vector_type(16))) _Float16 v16h;
typedef __attribute__((ext_vector_type(8)))  _Float16 v8h;
typedef __attribute__((ext_vector_type(8)))  float    v8f;

#define WMMA_F16(a, b, c) \
  __builtin_amdgcn_wmma_f32_16x16x32_f16(false, (a), false, (b), (short)0, (c), false, false)

// Fragment loader for CDNA5 16-bit WMMA operand layout (wave32).
// Lane L holds row (A) / column (B) = L%16; half = L/16 selects which K-chunks:
//   elements 0..7  = K[k0 + half*8 .. +8)      (VGPRs 0..3, f16 pairs)
//   elements 8..15 = K[k0 + 16 + half*8 .. +8) (VGPRs 4..7)
// Both are contiguous 16-byte runs -> two global_load_b128 per fragment.
static __device__ __forceinline__ v16h load_frag(const _Float16* __restrict__ base,
                                                 int ld, int row, int k0, int half) {
  const _Float16* p = base + (size_t)row * ld + k0 + half * 8;
  v8h lo = *(const v8h*)p;
  v8h hi = *(const v8h*)(p + 16);
  v16h r;
#pragma unroll
  for (int i = 0; i < 8; ++i) { r[i] = lo[i]; r[8 + i] = hi[i]; }
  return r;
}

// ---------------------------------------------------------------- convert f32 -> f16
// 8 elements/thread: two b128 loads, one b128 store. All sizes divisible by 2048.
__global__ __launch_bounds__(256) void cvt_f16_kernel(const float* __restrict__ src,
                                                      _Float16* __restrict__ dst, int n) {
  const int i = (blockIdx.x * 256 + threadIdx.x) * 8;
  if (i < n) {
    const float4 x0 = *(const float4*)(src + i);
    const float4 x1 = *(const float4*)(src + i + 4);
    v8h o;
    o[0] = (_Float16)x0.x; o[1] = (_Float16)x0.y; o[2] = (_Float16)x0.z; o[3] = (_Float16)x0.w;
    o[4] = (_Float16)x1.x; o[5] = (_Float16)x1.y; o[6] = (_Float16)x1.z; o[7] = (_Float16)x1.w;
    *(v8h*)(dst + i) = o;
  }
}

// ---------------------------------------------------------------- GEMM core
// Wave computes 32 rows x 64 cols (2 M-tiles x 4 N-tiles = 8 WMMA per K-step).
// Ping-pong double buffering (2x unrolled K loop, step 64): stage A computes
// with buffer 0 while loading buffer 1, stage B the reverse -> no copy movs.
static __device__ __forceinline__ void gemm_32x64_ping_pong(
    const _Float16* __restrict__ A, const _Float16* __restrict__ B,
    int m0, int n0, int kdim, int ln16, int half, v8f acc[2][4]) {
  v16h a0[2], b0[4], a1[2], b1[4];
  a0[0] = load_frag(A, kdim, m0 + ln16, 0, half);
  a0[1] = load_frag(A, kdim, m0 + 16 + ln16, 0, half);
#pragma unroll
  for (int t = 0; t < 4; ++t) b0[t] = load_frag(B, kdim, n0 + t * 16 + ln16, 0, half);

  for (int k0 = 0; k0 < kdim; k0 += 64) {
    const int k1 = k0 + 32;
    // stage A: load k1 into buffer 1, compute with buffer 0
    a1[0] = load_frag(A, kdim, m0 + ln16, k1, half);
    a1[1] = load_frag(A, kdim, m0 + 16 + ln16, k1, half);
#pragma unroll
    for (int t = 0; t < 4; ++t) b1[t] = load_frag(B, kdim, n0 + t * 16 + ln16, k1, half);
    __builtin_prefetch(A + (size_t)(m0 + ln16) * kdim + k1 + 32, 0, 1);
#pragma unroll
    for (int t = 0; t < 4; ++t) {
      acc[0][t] = WMMA_F16(a0[0], b0[t], acc[0][t]);
      acc[1][t] = WMMA_F16(a0[1], b0[t], acc[1][t]);
    }
    // stage B: load k0+64 into buffer 0, compute with buffer 1
    const int k2 = (k0 + 64 < kdim) ? k0 + 64 : 0;
    a0[0] = load_frag(A, kdim, m0 + ln16, k2, half);
    a0[1] = load_frag(A, kdim, m0 + 16 + ln16, k2, half);
#pragma unroll
    for (int t = 0; t < 4; ++t) b0[t] = load_frag(B, kdim, n0 + t * 16 + ln16, k2, half);
    __builtin_prefetch(B + (size_t)(n0 + ln16) * kdim + k2 + 32, 0, 1);
#pragma unroll
    for (int t = 0; t < 4; ++t) {
      acc[0][t] = WMMA_F16(a1[0], b1[t], acc[0][t]);
      acc[1][t] = WMMA_F16(a1[1], b1[t], acc[1][t]);
    }
  }
}

// ---------------------------------------------------------------- QKV GEMM + RoPE + norm
// Grid: (M/256, 3072/64). Block 256 = 8 waves. Block col range = one (t, head)
// slice -> RoPE + L2-norm close entirely in-register.
__global__ __launch_bounds__(256) void qkv_rope_norm_kernel(
    const _Float16* __restrict__ X,     // (4096 x 1024) tokens f16
    const _Float16* __restrict__ W,     // (3072 x 1024) qkv_w f16
    const float*    __restrict__ bias,  // (3072)
    _Float16* __restrict__ Q,           // (B*H, S, 64)
    _Float16* __restrict__ K,           // (B*H, S, 64)
    _Float16* __restrict__ Vt) {        // (B*H, 64, S)  transposed
  const int lane = threadIdx.x & 31;
  const int wave = threadIdx.x >> 5;
  const int half = lane >> 4;
  const int ln16 = lane & 15;
  const int m0 = blockIdx.x * 256 + wave * 32;   // global row (b*S + s)
  const int n0 = blockIdx.y * 64;                // global col in [0,3072)

  v8f acc[2][4] = {};
  gemm_32x64_ping_pong(X, W, m0, n0, 1024, ln16, half, acc);

  const int tsel = n0 / 1024;              // 0=q 1=k 2=v
  const int h    = (n0 % 1024) / 64;
  const int b    = m0 / 2048;
  const int bh   = b * 16 + h;

#pragma unroll
  for (int mi = 0; mi < 2; ++mi) {
    const int srow = (m0 % 2048) + mi * 16;  // sequence base of this tile's rows
    if (tsel == 2) {                         // V: bias + store transposed (d, s)
#pragma unroll
      for (int t = 0; t < 4; ++t) {
        const int d = t * 16 + ln16;
        const float bv = bias[n0 + t * 16 + ln16];
#pragma unroll
        for (int v = 0; v < 8; ++v) {
          const int s = srow + v + 8 * half;
          Vt[((size_t)bh * 64 + d) * 2048 + s] = (_Float16)(acc[mi][t][v] + bv);
        }
      }
    } else {                                 // Q/K: bias + RoPE + L2 normalize
      _Float16* dst = (tsel == 0) ? Q : K;
#pragma unroll
      for (int v = 0; v < 8; ++v) {
        const int s = srow + v + 8 * half;
        float vals[4], rv[4];
#pragma unroll
        for (int t = 0; t < 4; ++t) vals[t] = acc[mi][t][v] + bias[n0 + t * 16 + ln16];
        float ss = 0.f;
#pragma unroll
        for (int t = 0; t < 4; ++t) {
          const float partner = __shfl_xor(vals[t], 1);      // rope pair (same row)
          const int c = t * 16 + ln16;                       // in-head dim 0..63
          const float j = (float)(c >> 1);
          const float theta = (float)s * __expf(j * (-9.2103403719761836f / 32.f));
          float sn, cs;
          __sincosf(theta, &sn, &cs);
          rv[t] = (c & 1) ? (partner * sn + vals[t] * cs)    // odd:  xe*sin + xo*cos
                          : (vals[t] * cs - partner * sn);   // even: xe*cos - xo*sin
          ss += rv[t] * rv[t];
        }
        ss += __shfl_xor(ss, 1); ss += __shfl_xor(ss, 2);
        ss += __shfl_xor(ss, 4); ss += __shfl_xor(ss, 8);    // 16-lane row reduce
        const float scale = 1.f / fmaxf(sqrtf(ss), 1e-12f);
#pragma unroll
        for (int t = 0; t < 4; ++t)
          dst[((size_t)bh * 2048 + s) * 64 + t * 16 + ln16] = (_Float16)(rv[t] * scale);
      }
    }
  }
}

// ---------------------------------------------------------------- flash attention step
// One 32-key step: S^T = K*Q^T (C-layout puts queries in lanes -> per-lane softmax),
// lane-local repack of P^T into the B-fragment of O^T = V^T * P^T. Loads the NEXT
// step's K fragments (into kn*) right after the S^T WMMAs consume kc*, so they fly
// during the softmax VALU work and the PV WMMAs.
static __device__ __forceinline__ void flash_step(
    int s, int sn, const _Float16* __restrict__ Kb, const _Float16* __restrict__ Vb,
    const v16h& qb0, const v16h& qb1,
    v16h& kc0, v16h& kc1, v16h& kc2, v16h& kc3,
    v16h& kn0, v16h& kn1, v16h& kn2, v16h& kn3,
    v8f oT[4], float& mrun, float& lrun, int ln16, int half) {
  const float SCALE = 0.125f;  // 64^-0.5
  v8f sT0 = {}, sT1 = {};
  sT0 = WMMA_F16(kc0, qb0, sT0);
  sT0 = WMMA_F16(kc1, qb1, sT0);
  sT1 = WMMA_F16(kc2, qb0, sT1);
  sT1 = WMMA_F16(kc3, qb1, sT1);

  kn0 = load_frag(Kb, 64, sn + ln16, 0, half);
  kn1 = load_frag(Kb, 64, sn + ln16, 32, half);
  kn2 = load_frag(Kb, 64, sn + 16 + ln16, 0, half);
  kn3 = load_frag(Kb, 64, sn + 16 + ln16, 32, half);

  float mt = mrun;
#pragma unroll
  for (int v = 0; v < 8; ++v) {
    sT0[v] *= SCALE; sT1[v] *= SCALE;
    mt = fmaxf(mt, fmaxf(sT0[v], sT1[v]));
  }
  mt = fmaxf(mt, __shfl_xor(mt, 16));              // other half holds kv rows 8..15
  const float alpha = __expf(mrun - mt);
  mrun = mt;

  v16h pb;                                          // P^T as B-fragment (lane-local!)
  float colsum = 0.f;
#pragma unroll
  for (int v = 0; v < 8; ++v) {
    const float p0 = __expf(sT0[v] - mt);
    const float p1 = __expf(sT1[v] - mt);
    colsum += p0 + p1;
    pb[v] = (_Float16)p0;
    pb[8 + v] = (_Float16)p1;
  }
  colsum += __shfl_xor(colsum, 16);
  lrun = lrun * alpha + colsum;

#pragma unroll
  for (int t = 0; t < 4; ++t) {
#pragma unroll
    for (int v = 0; v < 8; ++v) oT[t][v] *= alpha;
    const v16h va = load_frag(Vb, 2048, t * 16 + ln16, s, half);  // V^T rows = d
    oT[t] = WMMA_F16(va, pb, oT[t]);
  }
}

// Grid: (S/128, B*H). Block 256 = 8 waves; wave owns 16 queries, streams K/V from
// the 192MB L2. kv loop 2x unrolled with ping-pong K-fragment buffers (no copies).
__global__ __launch_bounds__(256) void flash_attn_kernel(
    const _Float16* __restrict__ Q, const _Float16* __restrict__ K,
    const _Float16* __restrict__ Vt, _Float16* __restrict__ A2) {
  const int lane = threadIdx.x & 31;
  const int wave = threadIdx.x >> 5;
  const int half = lane >> 4;
  const int ln16 = lane & 15;
  const int bh = blockIdx.y;
  const int b = bh >> 4, h = bh & 15;
  const int q0 = blockIdx.x * 128 + wave * 16;

  const _Float16* Qb = Q  + (size_t)bh * 2048 * 64;
  const _Float16* Kb = K  + (size_t)bh * 2048 * 64;
  const _Float16* Vb = Vt + (size_t)bh * 64 * 2048;

  const v16h qb0 = load_frag(Qb, 64, q0 + ln16, 0, half);
  const v16h qb1 = load_frag(Qb, 64, q0 + ln16, 32, half);

  v8f oT[4] = {};
  float mrun = -3.0e30f, lrun = 0.f;

  v16h kA0 = load_frag(Kb, 64, ln16, 0, half);
  v16h kA1 = load_frag(Kb, 64, ln16, 32, half);
  v16h kA2 = load_frag(Kb, 64, 16 + ln16, 0, half);
  v16h kA3 = load_frag(Kb, 64, 16 + ln16, 32, half);
  v16h kB0, kB1, kB2, kB3;

  for (int s = 0; s < 2048; s += 64) {
    flash_step(s, s + 32, Kb, Vb, qb0, qb1,
               kA0, kA1, kA2, kA3, kB0, kB1, kB2, kB3,
               oT, mrun, lrun, ln16, half);
    const int sn = (s + 64 < 2048) ? s + 64 : 0;
    flash_step(s + 32, sn, Kb, Vb, qb0, qb1,
               kB0, kB1, kB2, kB3, kA0, kA1, kA2, kA3,
               oT, mrun, lrun, ln16, half);
  }

  const float inv = 1.f / lrun;
  const int q = q0 + ln16;
#pragma unroll
  for (int t = 0; t < 4; ++t)
#pragma unroll
    for (int v = 0; v < 8; ++v) {
      const int d = h * 64 + t * 16 + v + 8 * half;
      A2[((size_t)(b * 2048 + q)) * 1024 + d] = (_Float16)(oT[t][v] * inv);
    }
}

// ---------------------------------------------------------------- output projection
// Grid: (M/256, 1024/64). Ping-pong GEMM; bias + f32 store.
__global__ __launch_bounds__(256) void out_proj_kernel(
    const _Float16* __restrict__ A2, const _Float16* __restrict__ W,
    const float* __restrict__ bias, float* __restrict__ out) {
  const int lane = threadIdx.x & 31;
  const int wave = threadIdx.x >> 5;
  const int half = lane >> 4;
  const int ln16 = lane & 15;
  const int m0 = blockIdx.x * 256 + wave * 32;
  const int n0 = blockIdx.y * 64;

  v8f acc[2][4] = {};
  gemm_32x64_ping_pong(A2, W, m0, n0, 1024, ln16, half, acc);

#pragma unroll
  for (int mi = 0; mi < 2; ++mi)
#pragma unroll
    for (int t = 0; t < 4; ++t) {
      const float bv = bias[n0 + t * 16 + ln16];
#pragma unroll
      for (int v = 0; v < 8; ++v)
        out[(size_t)(m0 + mi * 16 + v + 8 * half) * 1024 + n0 + t * 16 + ln16] =
            acc[mi][t][v] + bv;
    }
}

// ---------------------------------------------------------------- launcher
extern "C" void kernel_launch(void* const* d_in, const int* in_sizes, int n_in,
                              void* d_out, int out_size, void* d_ws, size_t ws_size,
                              hipStream_t stream) {
  const float* tokens = (const float*)d_in[0];   // (2,2048,1024)
  const float* qkv_w  = (const float*)d_in[1];   // (3072,1024)
  const float* qkv_b  = (const float*)d_in[2];   // (3072)
  const float* out_w  = (const float*)d_in[3];   // (1024,1024)
  const float* out_b  = (const float*)d_in[4];   // (1024)
  float* out = (float*)d_out;

  char* ws = (char*)d_ws;
  const size_t MB = 1024ull * 1024ull;
  _Float16* Xf16 = (_Float16*)(ws + 0 * MB);    // 4M elems, 8MB
  _Float16* Wqkv = (_Float16*)(ws + 8 * MB);    // 3M elems, 6MB
  _Float16* Wout = (_Float16*)(ws + 14 * MB);   // 1M elems, 2MB
  _Float16* Qh   = (_Float16*)(ws + 16 * MB);   // 4M elems, 8MB
  _Float16* Kh   = (_Float16*)(ws + 24 * MB);   // 4M elems, 8MB
  _Float16* Vt   = (_Float16*)(ws + 32 * MB);   // 4M elems, 8MB
  _Float16* A2   = (_Float16*)(ws + 40 * MB);   // 4M elems, 8MB

  const int nTok = 2 * 2048 * 1024;
  const int nWq  = 3072 * 1024;
  const int nWo  = 1024 * 1024;
  cvt_f16_kernel<<<nTok / (256 * 8), 256, 0, stream>>>(tokens, Xf16, nTok);
  cvt_f16_kernel<<<nWq / (256 * 8), 256, 0, stream>>>(qkv_w, Wqkv, nWq);
  cvt_f16_kernel<<<nWo / (256 * 8), 256, 0, stream>>>(out_w, Wout, nWo);

  qkv_rope_norm_kernel<<<dim3(16, 48), 256, 0, stream>>>(Xf16, Wqkv, qkv_b, Qh, Kh, Vt);
  flash_attn_kernel<<<dim3(16, 32), 256, 0, stream>>>(Qh, Kh, Vt, A2);
  out_proj_kernel<<<dim3(16, 16), 256, 0, stream>>>(A2, Wout, out_b, out);
}
